// Hybrid_SelfAttention_MRHA3_31714038513971
// MI455X (gfx1250) — matrix operand order repaired and checked by reading, not compiled, not run on hardware
//
#include <hip/hip_runtime.h>
#include <hip/hip_bf16.h>
#include <math.h>

typedef __attribute__((ext_vector_type(8)))  float          v8f;
typedef __attribute__((ext_vector_type(16))) unsigned short v16u;
typedef __attribute__((ext_vector_type(16))) __bf16         v16bf;

#define CDIM  512
#define BATCH 8
#define TFULL 2048
#define THALF 1024

__device__ __forceinline__ unsigned short f2bf(float f) {
  unsigned int u = __builtin_bit_cast(unsigned int, f);
  unsigned int r = u + 0x7FFFu + ((u >> 16) & 1u);
  return (unsigned short)(r >> 16);
}
__device__ __forceinline__ float bf2f(unsigned short h) {
  unsigned int u = ((unsigned int)h) << 16;
  return __builtin_bit_cast(float, u);
}
__device__ __forceinline__ v16bf asbf(v16u v) { return __builtin_bit_cast(v16bf, v); }

// ---------------------------------------------------------------------------
// Transpose f32 weight [N,K] -> bf16 [K,N] so WMMA B-fragments are contiguous.
// ---------------------------------------------------------------------------
__global__ __launch_bounds__(256) void k_wtrans_bf16(const float* __restrict__ W,
                                                     unsigned short* __restrict__ WT) {
  int idx = blockIdx.x * 256 + threadIdx.x;   // 0 .. 512*512-1
  int k = idx >> 9, n = idx & 511;
  WT[idx] = f2bf(W[n * CDIM + k]);
}

// ---------------------------------------------------------------------------
// Strided conv1d downsample (stride 2, pad 1, kernel 3), NCH semantics.
// ---------------------------------------------------------------------------
__global__ __launch_bounds__(256) void k_conv_down(const float* __restrict__ x,
                                                   const float* __restrict__ cw,
                                                   const float* __restrict__ cb,
                                                   float* __restrict__ xd) {
  __shared__ __align__(16) float sx[3][CDIM];
  const int b = blockIdx.y, t2 = blockIdx.x;
  const int tc = 2 * t2;
  for (int i = threadIdx.x; i < 3 * CDIM; i += 256) {
    int kk = i / CDIM, c = i - kk * CDIM;
    int tt = tc - 1 + kk;
    sx[kk][c] = (tt >= 0 && tt < TFULL) ? x[((size_t)b * TFULL + tt) * CDIM + c] : 0.0f;
  }
  __syncthreads();
  for (int co = threadIdx.x; co < CDIM; co += 256) {
    const float* wp = cw + (size_t)co * CDIM * 3;
    float acc = cb[co];
    for (int ci = 0; ci < CDIM; ++ci)
      acc += sx[0][ci] * wp[ci * 3 + 0] + sx[1][ci] * wp[ci * 3 + 1] + sx[2][ci] * wp[ci * 3 + 2];
    xd[((size_t)b * THALF + t2) * CDIM + co] = acc;
  }
}

// ---------------------------------------------------------------------------
// LayerNorm f32 -> bf16 (one row per block).
// ---------------------------------------------------------------------------
__global__ __launch_bounds__(256) void k_layernorm_bf16(const float* __restrict__ x,
                                                        const float* __restrict__ w,
                                                        const float* __restrict__ bb,
                                                        unsigned short* __restrict__ y) {
  __shared__ float r1[8], r2[8], stat[2];
  const int row = blockIdx.x;
  const float* xr = x + (size_t)row * CDIM;
  int tid = threadIdx.x, lane = tid & 31, wv = tid >> 5;
  float s = 0.f, s2 = 0.f;
  for (int c = tid; c < CDIM; c += 256) { float v = xr[c]; s += v; s2 += v * v; }
#pragma unroll
  for (int off = 16; off > 0; off >>= 1) {
    s += __shfl_xor(s, off, 32);
    s2 += __shfl_xor(s2, off, 32);
  }
  if (lane == 0) { r1[wv] = s; r2[wv] = s2; }
  __syncthreads();
  if (tid == 0) {
    float a = 0.f, b2 = 0.f;
#pragma unroll
    for (int i = 0; i < 8; ++i) { a += r1[i]; b2 += r2[i]; }
    float mean = a / (float)CDIM;
    float var  = b2 / (float)CDIM - mean * mean;
    stat[0] = mean;
    stat[1] = rsqrtf(var + 1e-5f);
  }
  __syncthreads();
  float mean = stat[0], rstd = stat[1];
  for (int c = tid; c < CDIM; c += 256)
    y[(size_t)row * CDIM + c] = f2bf((xr[c] - mean) * rstd * w[c] + bb[c]);
}

// ---------------------------------------------------------------------------
// Row-wise L2 normalization in-place on bf16 rows (cosine branch).
// ---------------------------------------------------------------------------
__global__ __launch_bounds__(256) void k_rownorm_bf16(unsigned short* __restrict__ q) {
  __shared__ float r1[8];
  __shared__ float sInv;
  const int row = blockIdx.x;
  unsigned short* qr = q + (size_t)row * CDIM;
  int tid = threadIdx.x, lane = tid & 31, wv = tid >> 5;
  float s = 0.f;
  for (int c = tid; c < CDIM; c += 256) { float v = bf2f(qr[c]); s += v * v; }
#pragma unroll
  for (int off = 16; off > 0; off >>= 1) s += __shfl_xor(s, off, 32);
  if (lane == 0) r1[wv] = s;
  __syncthreads();
  if (tid == 0) {
    float a = 0.f;
#pragma unroll
    for (int i = 0; i < 8; ++i) a += r1[i];
    sInv = 1.0f / (sqrtf(a) + 1e-8f);
  }
  __syncthreads();
  float iv = sInv;
  for (int c = tid; c < CDIM; c += 256) qr[c] = f2bf(bf2f(qr[c]) * iv);
}

// ---------------------------------------------------------------------------
// WMMA GEMM: Y[M,512] = A[M,512] @ W^T + bias, bf16 in / f32 acc / bf16 out.
// One 16x16 output tile per wave; K swept 512 in 16 v_wmma_f32_16x16x32_bf16.
// ---------------------------------------------------------------------------
__global__ __launch_bounds__(256) void k_gemm_bf16(const unsigned short* __restrict__ A,
                                                   const unsigned short* __restrict__ WT,
                                                   const float* __restrict__ bias,
                                                   unsigned short* __restrict__ Y,
                                                   int M) {
  const int NT = CDIM / 16;  // 32 n-tiles
  int wid  = blockIdx.x * (blockDim.x >> 5) + (threadIdx.x >> 5);
  int lane = threadIdx.x & 31;
  int tm = wid / NT, tn = wid - tm * NT;
  int m0 = tm * 16, n0 = tn * 16;
  if (m0 >= M) return;
  int mr = lane & 15, hh = lane >> 4;
  v8f acc = {0.f, 0.f, 0.f, 0.f, 0.f, 0.f, 0.f, 0.f};
  const unsigned short* Arow = A + (size_t)(m0 + mr) * CDIM;
  for (int k0 = 0; k0 < CDIM; k0 += 32) {
    v16u a, b;
#pragma unroll
    for (int v = 0; v < 8; ++v) {  // A frag: 16-bit 16x32 layout
      int k = k0 + ((v < 4) ? 0 : 16) + hh * 8 + (v & 3) * 2;
      unsigned int p = *(const unsigned int*)(Arow + k);
      a[2 * v] = (unsigned short)p; a[2 * v + 1] = (unsigned short)(p >> 16);
    }
#pragma unroll
    for (int v = 0; v < 8; ++v) {  // B frag: lane = n, elements = k (same order as A)
      int k = k0 + ((v < 4) ? 0 : 16) + hh * 8 + (v & 3) * 2;
      b[2 * v] = WT[(size_t)k * CDIM + n0 + mr]; b[2 * v + 1] = WT[(size_t)(k + 1) * CDIM + n0 + mr];
    }
    acc = __builtin_amdgcn_wmma_f32_16x16x32_bf16(false, asbf(a), false, asbf(b),
                                                  (short)0, acc, false, false);
  }
  float bn = bias[n0 + mr];
#pragma unroll
  for (int r = 0; r < 8; ++r) {
    int m = m0 + r + 8 * hh;
    Y[(size_t)m * CDIM + n0 + mr] = f2bf(acc[r] + bn);
  }
}

// ---------------------------------------------------------------------------
// Flash attention, bf16 WMMA, head dim 512, online softmax.
// Block = (16 query rows, 1 batch), 8 waves; wave w owns channels [64w,64w+64).
// QK^T k-reduction distributed across waves, reduced via LDS; P*V per-wave.
// scale from const, or 1/max(*tempPtr,0.01) for the cosine branch.
// ---------------------------------------------------------------------------
__global__ __launch_bounds__(256) void k_flash_bf16(const unsigned short* __restrict__ Q,
                                                    const unsigned short* __restrict__ K,
                                                    const unsigned short* __restrict__ V,
                                                    float* __restrict__ O,
                                                    int T, int S, float scaleConst,
                                                    const float* __restrict__ tempPtr) {
  __shared__ __align__(16) unsigned short sTile[32 * CDIM];  // K^T then V  (32KB)
  __shared__ __align__(16) float sSp[8][16][32];             // partial S   (16KB)
  __shared__ __align__(16) unsigned short sP[16][32];        // P tile bf16 (1KB)
  __shared__ float sAlpha[16], sRowMax[16], sRowSum[16];

  float scale = scaleConst;
  if (tempPtr) scale = 1.0f / fmaxf(tempPtr[0], 0.01f);

  const int b  = blockIdx.y;
  const int q0 = blockIdx.x * 16;
  const unsigned short* Qb = Q + ((size_t)b * T) * CDIM;
  const unsigned short* Kb = K + ((size_t)b * S) * CDIM;
  const unsigned short* Vb = V + ((size_t)b * S) * CDIM;
  float* Ob = O + ((size_t)b * T) * CDIM;

  const int tid = threadIdx.x;
  const int wv = tid >> 5, lane = tid & 31;
  const int mr = lane & 15, hh = lane >> 4;

  // Preload the wave's Q A-fragments (channels [64*wv, 64*wv+64)), reused all tiles.
  v16u aQ0, aQ1;
  {
    const unsigned short* Qr = Qb + (size_t)(q0 + mr) * CDIM + wv * 64;
#pragma unroll
    for (int v = 0; v < 8; ++v) {
      int kof = ((v < 4) ? 0 : 16) + hh * 8 + (v & 3) * 2;
      unsigned int p0 = *(const unsigned int*)(Qr + kof);
      aQ0[2 * v] = (unsigned short)p0; aQ0[2 * v + 1] = (unsigned short)(p0 >> 16);
      unsigned int p1 = *(const unsigned int*)(Qr + 32 + kof);
      aQ1[2 * v] = (unsigned short)p1; aQ1[2 * v + 1] = (unsigned short)(p1 >> 16);
    }
  }

  if (tid < 16) { sRowMax[tid] = -1e30f; sRowSum[tid] = 0.0f; sAlpha[tid] = 1.0f; }

  v8f acc[4];
#pragma unroll
  for (int t = 0; t < 4; ++t)
#pragma unroll
    for (int r = 0; r < 8; ++r) acc[t][r] = 0.0f;

  for (int s0 = 0; s0 < S; s0 += 32) {
    __syncthreads();  // previous iteration's V reads complete before refill
    // Stage K tile TRANSPOSED: sTile[c*32 + key]  (coalesced global, strided LDS)
    for (int idx = tid; idx < 32 * 256; idx += 256) {
      int key = idx >> 8;
      int cp  = (idx & 255) * 2;
      unsigned int p = *(const unsigned int*)(Kb + (size_t)(s0 + key) * CDIM + cp);
      sTile[(cp)     * 32 + key] = (unsigned short)p;
      sTile[(cp + 1) * 32 + key] = (unsigned short)(p >> 16);
    }
    __syncthreads();

    // Partial S = Q_slice @ K_slice^T over this wave's 64 channels.
    v8f sa[2];
#pragma unroll
    for (int h = 0; h < 2; ++h)
#pragma unroll
      for (int r = 0; r < 8; ++r) sa[h][r] = 0.0f;
#pragma unroll
    for (int h = 0; h < 2; ++h) {     // key half (n0 = 16h)
#pragma unroll
      for (int kk = 0; kk < 2; ++kk) {  // 32-channel k-step
        int cb = wv * 64 + kk * 32;
        v16u bk;
#pragma unroll
        for (int v = 0; v < 8; ++v) {
          int kof = ((v < 4) ? 0 : 16) + hh * 8 + (v & 3) * 2;
          bk[2 * v] = sTile[(cb + kof) * 32 + h * 16 + mr]; bk[2 * v + 1] = sTile[(cb + kof + 1) * 32 + h * 16 + mr];
        }
        sa[h] = __builtin_amdgcn_wmma_f32_16x16x32_bf16(false, asbf(kk ? aQ1 : aQ0),
                                                        false, asbf(bk),
                                                        (short)0, sa[h], false, false);
      }
    }
#pragma unroll
    for (int h = 0; h < 2; ++h)
#pragma unroll
      for (int r = 0; r < 8; ++r)
        sSp[wv][r + 8 * hh][h * 16 + mr] = sa[h][r];
    __syncthreads();

    // Cross-wave reduce + online softmax (threads 0..15: one query row each),
    // overlapped with the V-tile fill by all 256 threads.
    if (tid < 16) {
      float vals[32];
      float tmax = -1e30f;
#pragma unroll
      for (int j = 0; j < 32; ++j) {
        float s = 0.0f;
#pragma unroll
        for (int w = 0; w < 8; ++w) s += sSp[w][tid][j];
        s *= scale;
        vals[j] = s;
        tmax = fmaxf(tmax, s);
      }
      float mold = sRowMax[tid];
      float mnew = fmaxf(mold, tmax);
      float alpha = __expf(mold - mnew);
      float lsum = 0.0f;
#pragma unroll
      for (int j = 0; j < 32; ++j) {
        float p = __expf(vals[j] - mnew);
        lsum += p;
        sP[tid][j] = f2bf(p);
      }
      sRowSum[tid] = sRowSum[tid] * alpha + lsum;
      sRowMax[tid] = mnew;
      sAlpha[tid]  = alpha;
    }
    for (int idx = tid; idx < 32 * 256; idx += 256) {  // V tile, row-major
      int key = idx >> 8;
      int cp  = (idx & 255) * 2;
      *(unsigned int*)(&sTile[key * CDIM + cp]) =
          *(const unsigned int*)(Vb + (size_t)(s0 + key) * CDIM + cp);
    }
    __syncthreads();

    // P A-fragment (16x32 bf16 from LDS) then rescale + P@V per 16-col tile.
    v16u aP;
#pragma unroll
    for (int v = 0; v < 8; ++v) {
      int kof = ((v < 4) ? 0 : 16) + hh * 8 + (v & 3) * 2;
      unsigned int p = *(const unsigned int*)(&sP[mr][kof]);
      aP[2 * v] = (unsigned short)p; aP[2 * v + 1] = (unsigned short)(p >> 16);
    }
#pragma unroll
    for (int t = 0; t < 4; ++t) {
      int c0 = wv * 64 + t * 16;
      v16u bv;
#pragma unroll
      for (int v = 0; v < 8; ++v) {  // B[k=key per element][n=channel=lane%16]
        int kof = ((v < 4) ? 0 : 16) + hh * 8 + (v & 3) * 2;
        bv[2 * v] = sTile[kof * CDIM + c0 + mr]; bv[2 * v + 1] = sTile[(kof + 1) * CDIM + c0 + mr];
      }
#pragma unroll
      for (int r = 0; r < 8; ++r) acc[t][r] *= sAlpha[r + 8 * hh];
      acc[t] = __builtin_amdgcn_wmma_f32_16x16x32_bf16(false, asbf(aP), false, asbf(bv),
                                                       (short)0, acc[t], false, false);
    }
  }
  __syncthreads();
#pragma unroll
  for (int t = 0; t < 4; ++t) {
    int c0 = wv * 64 + t * 16;
#pragma unroll
    for (int r = 0; r < 8; ++r) {
      int m = r + 8 * hh;
      Ob[(size_t)(q0 + m) * CDIM + c0 + mr] = acc[t][r] / sRowSum[m];
    }
  }
}

// ---------------------------------------------------------------------------
// Gated fusion: logits = [oc|od|ocs] @ gate_w^T + gate_b; softmax over 3; mix.
// ---------------------------------------------------------------------------
__global__ __launch_bounds__(256) void k_fuse_gate(const float* __restrict__ OC,
                                                   const float* __restrict__ OD,
                                                   const float* __restrict__ OS,
                                                   const float* __restrict__ gw,
                                                   const float* __restrict__ gb,
                                                   float* __restrict__ out) {
  __shared__ float r0[8], r1[8], r2[8], g[3];
  const int row = blockIdx.x;
  const float* a = OC + (size_t)row * CDIM;
  const float* d = OD + (size_t)row * CDIM;
  const float* e = OS + (size_t)row * CDIM;
  int tid = threadIdx.x, lane = tid & 31, wv = tid >> 5;
  float p0 = 0.f, p1 = 0.f, p2 = 0.f;
  for (int c = tid; c < CDIM; c += 256) {
    float va = a[c], vd = d[c], ve = e[c];
    p0 += va * gw[c]            + vd * gw[512 + c]        + ve * gw[1024 + c];
    p1 += va * gw[1536 + c]     + vd * gw[1536 + 512 + c] + ve * gw[1536 + 1024 + c];
    p2 += va * gw[3072 + c]     + vd * gw[3072 + 512 + c] + ve * gw[3072 + 1024 + c];
  }
#pragma unroll
  for (int off = 16; off > 0; off >>= 1) {
    p0 += __shfl_xor(p0, off, 32);
    p1 += __shfl_xor(p1, off, 32);
    p2 += __shfl_xor(p2, off, 32);
  }
  if (lane == 0) { r0[wv] = p0; r1[wv] = p1; r2[wv] = p2; }
  __syncthreads();
  if (tid == 0) {
    float l0 = gb[0], l1 = gb[1], l2 = gb[2];
#pragma unroll
    for (int i = 0; i < 8; ++i) { l0 += r0[i]; l1 += r1[i]; l2 += r2[i]; }
    float mx = fmaxf(l0, fmaxf(l1, l2));
    float e0 = __expf(l0 - mx), e1 = __expf(l1 - mx), e2 = __expf(l2 - mx);
    float inv = 1.0f / (e0 + e1 + e2);
    g[0] = e0 * inv; g[1] = e1 * inv; g[2] = e2 * inv;
  }
  __syncthreads();
  float g0 = g[0], g1 = g[1], g2 = g[2];
  for (int c = tid; c < CDIM; c += 256)
    out[(size_t)row * CDIM + c] = g0 * a[c] + g1 * d[c] + g2 * e[c];
}

// ---------------------------------------------------------------------------
extern "C" void kernel_launch(void* const* d_in, const int* in_sizes, int n_in,
                              void* d_out, int out_size, void* d_ws, size_t ws_size,
                              hipStream_t stream) {
  (void)in_sizes; (void)n_in; (void)out_size; (void)ws_size;
  const float* x      = (const float*)d_in[0];
  const float* conv_w = (const float*)d_in[1];
  const float* conv_b = (const float*)d_in[2];
  const float* lnc_w  = (const float*)d_in[3];
  const float* lnc_b  = (const float*)d_in[4];
  const float* lnd_w  = (const float*)d_in[5];
  const float* lnd_b  = (const float*)d_in[6];
  const float* lns_w  = (const float*)d_in[7];
  const float* lns_b  = (const float*)d_in[8];
  const float* qcw = (const float*)d_in[9];  const float* qcb = (const float*)d_in[10];
  const float* kcw = (const float*)d_in[11]; const float* kcb = (const float*)d_in[12];
  const float* vcw = (const float*)d_in[13]; const float* vcb = (const float*)d_in[14];
  const float* qdw = (const float*)d_in[15]; const float* qdb = (const float*)d_in[16];
  const float* kdw = (const float*)d_in[17]; const float* kdb = (const float*)d_in[18];
  const float* vdw = (const float*)d_in[19]; const float* vdb = (const float*)d_in[20];
  const float* qsw = (const float*)d_in[21]; const float* qsb = (const float*)d_in[22];
  const float* ksw = (const float*)d_in[23]; const float* ksb = (const float*)d_in[24];
  const float* vsw = (const float*)d_in[25]; const float* vsb = (const float*)d_in[26];
  const float* cos_temp = (const float*)d_in[27];
  const float* gate_w   = (const float*)d_in[28];
  const float* gate_b   = (const float*)d_in[29];

  char* ws = (char*)d_ws;
  const size_t BTC = (size_t)BATCH * TFULL * CDIM;  // 8,388,608 elements
  float*          outC = (float*)(ws);                          // f32, BTC
  float*          outD = (float*)(ws + BTC * 4);                // f32, BTC
  float*          outS = (float*)(ws + BTC * 8);                // f32, BTC
  float*          xd   = (float*)(ws + BTC * 12);               // f32, BTC/2
  unsigned short* xn   = (unsigned short*)(ws + BTC * 14);      // bf16, BTC
  unsigned short* xdn  = (unsigned short*)(ws + BTC * 16);      // bf16, BTC/2
  unsigned short* qb   = (unsigned short*)(ws + BTC * 17);      // bf16, BTC
  unsigned short* kb   = (unsigned short*)(ws + BTC * 19);      // bf16, BTC
  unsigned short* vb   = (unsigned short*)(ws + BTC * 21);      // bf16, BTC
  unsigned short* wt   = (unsigned short*)(ws + BTC * 23);      // bf16, 9*512*512

  const float scale = 0.044194173824159216f;  // 1/sqrt(512)
  const size_t WSZ = (size_t)CDIM * CDIM;     // 262144

  // Weights -> transposed bf16 (order: qc kc vc qd kd vd qs ks vs)
  const float* wsrc[9] = {qcw, kcw, vcw, qdw, kdw, vdw, qsw, ksw, vsw};
  for (int i = 0; i < 9; ++i)
    k_wtrans_bf16<<<1024, 256, 0, stream>>>(wsrc[i], wt + (size_t)i * WSZ);

  k_conv_down<<<dim3(THALF, BATCH), 256, 0, stream>>>(x, conv_w, conv_b, xd);

  const int Mf = BATCH * TFULL;                      // 16384
  const int Mh = BATCH * THALF;                      // 8192
  const int gemmBlkF = (Mf / 16) * (CDIM / 16) / 8;  // 4096 blocks (8 waves each)
  const int gemmBlkH = (Mh / 16) * (CDIM / 16) / 8;  // 2048

  // ---- cross-resolution branch ----
  k_layernorm_bf16<<<Mf, 256, 0, stream>>>(x,  lnc_w, lnc_b, xn);
  k_layernorm_bf16<<<Mh, 256, 0, stream>>>(xd, lnc_w, lnc_b, xdn);
  k_gemm_bf16<<<gemmBlkF, 256, 0, stream>>>(xn,  wt + 0 * WSZ, qcb, qb, Mf);
  k_gemm_bf16<<<gemmBlkH, 256, 0, stream>>>(xdn, wt + 1 * WSZ, kcb, kb, Mh);
  k_gemm_bf16<<<gemmBlkH, 256, 0, stream>>>(xdn, wt + 2 * WSZ, vcb, vb, Mh);
  k_flash_bf16<<<dim3(TFULL / 16, BATCH), 256, 0, stream>>>(qb, kb, vb, outC,
                                                            TFULL, THALF, scale, nullptr);

  // ---- scaled dot-product branch ----
  k_layernorm_bf16<<<Mf, 256, 0, stream>>>(x, lnd_w, lnd_b, xn);
  k_gemm_bf16<<<gemmBlkF, 256, 0, stream>>>(xn, wt + 3 * WSZ, qdb, qb, Mf);
  k_gemm_bf16<<<gemmBlkF, 256, 0, stream>>>(xn, wt + 4 * WSZ, kdb, kb, Mf);
  k_gemm_bf16<<<gemmBlkF, 256, 0, stream>>>(xn, wt + 5 * WSZ, vdb, vb, Mf);
  k_flash_bf16<<<dim3(TFULL / 16, BATCH), 256, 0, stream>>>(qb, kb, vb, outD,
                                                            TFULL, TFULL, scale, nullptr);

  // ---- cosine-similarity branch ----
  k_layernorm_bf16<<<Mf, 256, 0, stream>>>(x, lns_w, lns_b, xn);
  k_gemm_bf16<<<gemmBlkF, 256, 0, stream>>>(xn, wt + 6 * WSZ, qsb, qb, Mf);
  k_gemm_bf16<<<gemmBlkF, 256, 0, stream>>>(xn, wt + 7 * WSZ, ksb, kb, Mf);
  k_gemm_bf16<<<gemmBlkF, 256, 0, stream>>>(xn, wt + 8 * WSZ, vsb, vb, Mf);
  k_rownorm_bf16<<<Mf, 256, 0, stream>>>(qb);
  k_rownorm_bf16<<<Mf, 256, 0, stream>>>(kb);
  k_flash_bf16<<<dim3(TFULL / 16, BATCH), 256, 0, stream>>>(qb, kb, vb, outS,
                                                            TFULL, TFULL, 1.0f, cos_temp);

  // ---- gated fusion ----
  k_fuse_gate<<<Mf, 256, 0, stream>>>(outC, outD, outS, gate_w, gate_b, (float*)d_out);
}
